// Feature_RNNLM_27943057228461
// MI455X (gfx1250) — matrix operand order repaired and checked by reading, not compiled, not run on hardware
//
#include <hip/hip_runtime.h>
#include <hip/hip_bf16.h>
#include <math.h>

// Problem constants (fixed by reference setup_inputs).
static constexpr int Bc = 64, Tc = 128, Vc = 16000, Dc = 64, Hc = 512;
static constexpr int N4 = 4 * Hc;        // 2048
static constexpr int MBT = Bc * Tc;      // 8192

typedef __attribute__((ext_vector_type(16))) _Float16 v16h;
typedef __attribute__((ext_vector_type(8)))  _Float16 v8h;
typedef __attribute__((ext_vector_type(8)))  float    v8f;

#define DEV static __device__ __forceinline__

// ---------------------------------------------------------------------------
// A-fragment loader: 16x32 f16 tile from row-major A[m][k] (contiguous k = 16*(lane/16) + elem)
// ---------------------------------------------------------------------------
DEV v16h load_a16x32(const _Float16* A, int lda, int m0, int k0, int lane) {
  int m  = m0 + (lane & 15);
  int kb = k0 + ((lane >> 4) << 4);
  const _Float16* p = A + (size_t)m * lda + kb;
  v8h lo = *(const v8h*)(p);
  v8h hi = *(const v8h*)(p + 8);
  v16h r;
#pragma unroll
  for (int i = 0; i < 8; ++i) { r[i] = lo[i]; r[8 + i] = hi[i]; }
  return r;
}

DEV v8f wmma_f16(v16h a, v16h b, v8f c) {
  return __builtin_amdgcn_wmma_f32_16x16x32_f16(false, a, false, b, (short)0,
                                                c, false, false);
}

// Fast gate math on v_exp_f32 / v_rcp_f32 (critical path of the recurrence).
DEV float fsig(float x)  { return __builtin_amdgcn_rcpf(1.0f + __expf(-x)); }
DEV float ftanh(float x) { return 1.0f - 2.0f * __builtin_amdgcn_rcpf(1.0f + __expf(2.0f * x)); }

// ---------------------------------------------------------------------------
// Weight packer: W[N][K] f32 row-major -> f16 B-fragments.
// Block (nt,kt) = 32 lanes x 16 halfs (1KB), block index = nt*(K/32)+kt.
// lane = (n&15) + 16*((k&31)>>4), elem = k&15.
// ---------------------------------------------------------------------------
__global__ void pack_w_kernel(const float* __restrict__ W,
                              _Float16* __restrict__ out, int N, int K) {
  int idx = blockIdx.x * 256 + threadIdx.x;
  if (idx >= N * K) return;
  int n = idx / K, k = idx % K;
  int nt = n >> 4, kt = k >> 5;
  int lane = (n & 15) + (((k & 31) >> 4) << 4);
  int e = k & 15;
  int kB = K >> 5;
  out[((size_t)(nt * kB + kt) << 9) + lane * 16 + e] = (_Float16)W[idx];
}

__global__ void add_bias_kernel(const float* __restrict__ a,
                                const float* __restrict__ b,
                                float* __restrict__ o, int n) {
  int i = blockIdx.x * 256 + threadIdx.x;
  if (i < n) o[i] = a[i] + b[i];
}

// Embedding gather + cast, time-major layout x16[t][b][d].
__global__ void embed_cast_kernel(const int* __restrict__ batch,
                                  const float* __restrict__ feat,
                                  _Float16* __restrict__ x16) {
  int idx = blockIdx.x * 256 + threadIdx.x;
  if (idx >= Tc * Bc * Dc) return;
  int d = idx % Dc;
  int tb = idx / Dc;
  int b = tb % Bc, t = tb / Bc;
  int tok = batch[b * Tc + t];
  x16[idx] = (_Float16)feat[(size_t)tok * Dc + d];
}

// ---------------------------------------------------------------------------
// Bulk GEMM:  C[m][n] = sum_k A[m][k]*W[n][k] + bias[n]
// 8 waves/WG, wave tile 32x64 (8 accumulators), register double-buffered.
// remap!=0 scatters time-major rows (t*B+b) to batch-major rows (b*T+t).
// ---------------------------------------------------------------------------
__global__ void wmma_gemm_bias_kernel(const _Float16* __restrict__ A, int lda,
                                      const _Float16* __restrict__ Wp,
                                      const float* __restrict__ bias,
                                      float* __restrict__ C,
                                      int N, int K, int remap) {
  int lane = threadIdx.x & 31;
  int wave = threadIdx.x >> 5;
  int m0 = blockIdx.y * 256 + wave * 32;
  int n0 = blockIdx.x * 64;
  int kB = K >> 5;
  v8f acc[2][4] = {};

  auto loadB = [&](int j, int kt) -> v16h {
    size_t nt = (size_t)((n0 >> 4) + j);
    return *(const v16h*)(Wp + ((nt * kB + kt) << 9) + lane * 16);
  };

  v16h ca0 = load_a16x32(A, lda, m0, 0, lane);
  v16h ca1 = load_a16x32(A, lda, m0 + 16, 0, lane);
  v16h cb[4];
#pragma unroll
  for (int j = 0; j < 4; ++j) cb[j] = loadB(j, 0);

#pragma unroll 2
  for (int kt = 0; kt + 1 < kB; ++kt) {
    // issue next-iteration fragment loads first (overlap with WMMAs below)
    v16h na0 = load_a16x32(A, lda, m0, (kt + 1) << 5, lane);
    v16h na1 = load_a16x32(A, lda, m0 + 16, (kt + 1) << 5, lane);
    v16h nb[4];
#pragma unroll
    for (int j = 0; j < 4; ++j) nb[j] = loadB(j, kt + 1);
    if (kt + 2 < kB)  // stream weight blocks ahead (global_prefetch_b8)
      __builtin_prefetch(Wp + (((size_t)(n0 >> 4) * kB + kt + 2) << 9), 0, 1);
#pragma unroll
    for (int j = 0; j < 4; ++j) {
      acc[0][j] = wmma_f16(ca0, cb[j], acc[0][j]);
      acc[1][j] = wmma_f16(ca1, cb[j], acc[1][j]);
    }
    ca0 = na0; ca1 = na1;
#pragma unroll
    for (int j = 0; j < 4; ++j) cb[j] = nb[j];
  }
#pragma unroll
  for (int j = 0; j < 4; ++j) {
    acc[0][j] = wmma_f16(ca0, cb[j], acc[0][j]);
    acc[1][j] = wmma_f16(ca1, cb[j], acc[1][j]);
  }

  int col = lane & 15;
  int rbase = (lane >> 4) << 3;
#pragma unroll
  for (int j = 0; j < 4; ++j) {
    int n = n0 + j * 16 + col;
    float bv = bias ? bias[n] : 0.0f;
#pragma unroll
    for (int mt = 0; mt < 2; ++mt) {
#pragma unroll
      for (int v = 0; v < 8; ++v) {
        int m = m0 + mt * 16 + rbase + v;
        int om = m;
        if (remap) { int tt = m / Bc, bb = m % Bc; om = bb * Tc + tt; }
        C[(size_t)om * N + n] = acc[mt][j][v] + bv;
      }
    }
  }
}

// ---------------------------------------------------------------------------
// Device-scope grid barrier (persistent scan kernel, all WGs resident).
// ---------------------------------------------------------------------------
DEV void gsync(unsigned* cnt, unsigned target) {
  __threadfence();
  __syncthreads();
  if (threadIdx.x == 0) {
    atomicAdd(cnt, 1u);
    while (__hip_atomic_load(cnt, __ATOMIC_RELAXED, __HIP_MEMORY_SCOPE_AGENT) <
           target)
      __builtin_amdgcn_s_sleep(1);
  }
  __syncthreads();
  __threadfence();
}

// ---------------------------------------------------------------------------
// Persistent, fully-fused LSTM scan for one layer (the latency-bound part).
// Grid = 32 WGs x 128 thr = 128 waves.  Wave (mt,jt) owns the h/c slice
// rows mt*16..+15, cols jt*16..+15 and computes ALL FOUR gate tiles for it
// (n = jt*16 + q*512, q=0..3).  Consequences:
//   - gate nonlinearity is pure register math (accumulator fragments of the
//     four gates are element-aligned), no gbuf, no second barrier
//   - c lives in 8 VGPRs for the entire 128-step scan
//   - h is ping-pong buffered (write h(t+1) to other parity) -> the single
//     end-of-step barrier orders all reads of h(t) before any write of h(t+2)
//   - the wave's 64 Whh B-fragments (4 gates x 16 kt) are loop-invariant and
//     hoisted into 512 VGPRs (CDNA5: up to 1024 VGPRs/wave, 1 wave/SIMD here)
// Per-step critical path: 32 xg loads + 16 h-fragment loads + 64 WMMAs +
// ~40 trans VALU + 16 stores + 1 grid barrier.
// ---------------------------------------------------------------------------
__global__ void __launch_bounds__(128, 1)
lstm_scan_kernel(const float* __restrict__ xg,
                 const _Float16* __restrict__ WhhP,
                 float* cbuf,             // [B][H] f32 carried state
                 _Float16* hpp,           // [2][B][H] f16 ping-pong h
                 _Float16* __restrict__ yout,  // [T][B][H] f16
                 int initZero, unsigned* cnt) {
  const int NWG = gridDim.x;                  // 32
  int lane = threadIdx.x & 31;
  int wave = threadIdx.x >> 5;
  int gid = blockIdx.x * 128 + threadIdx.x;   // 0..4095
  unsigned gen = 0;

  int tile = blockIdx.x * 4 + wave;           // 0..127
  int mt = tile & 3, jt = tile >> 2;          // 4 x 32 slices of [64][512]
  int m0 = mt << 4, j0 = jt << 4;
  int col = lane & 15;
  int rbase = (lane >> 4) << 3;

  // Hoist this wave's recurrent-weight fragments: 4 gates x 16 kt x 8 VGPR.
  v16h bfr[4][16];
#pragma unroll
  for (int q = 0; q < 4; ++q)
#pragma unroll
    for (int kt = 0; kt < 16; ++kt)
      bfr[q][kt] = *(const v16h*)(
          WhhP + (((size_t)(q * 32 + jt) * 16 + kt) << 9) + lane * 16);

  // Cell state for this wave's slice, register-resident across the scan.
  v8f creg;
  if (initZero) {
#pragma unroll
    for (int v = 0; v < 8; ++v) creg[v] = 0.0f;
    for (int i = gid; i < Bc * Hc; i += NWG * 128) hpp[i] = (_Float16)0.0f;
  } else {
#pragma unroll
    for (int v = 0; v < 8; ++v)
      creg[v] = cbuf[(size_t)(m0 + rbase + v) * Hc + (j0 + col)];
  }
  ++gen; gsync(cnt, gen * NWG);

  for (int t = 0; t < Tc; ++t) {
    const _Float16* hsrc = hpp + (size_t)(t & 1) * (Bc * Hc);
    _Float16* hdst = hpp + (size_t)((t + 1) & 1) * (Bc * Hc);

    // ---- four gate tiles: acc[q] = xg[t] slice + h @ Whh^T ----
    v8f acc[4];
    const float* xr = xg + (size_t)t * Bc * N4;
#pragma unroll
    for (int q = 0; q < 4; ++q)
#pragma unroll
      for (int v = 0; v < 8; ++v)
        acc[q][v] = xr[(size_t)(m0 + rbase + v) * N4 + (q * Hc + j0 + col)];
#pragma unroll
    for (int kt = 0; kt < 16; ++kt) {
      v16h a = load_a16x32(hsrc, Hc, m0, kt << 5, lane);
#pragma unroll
      for (int q = 0; q < 4; ++q) acc[q] = wmma_f16(a, bfr[q][kt], acc[q]);
    }

    // ---- gates in registers (i,f,g,o), update c, emit h ----
#pragma unroll
    for (int v = 0; v < 8; ++v) {
      float si = fsig(acc[0][v]);
      float sf = fsig(acc[1][v]);
      float tg = ftanh(acc[2][v]);
      float so = fsig(acc[3][v]);
      float c = sf * creg[v] + si * tg;
      creg[v] = c;
      _Float16 hh = (_Float16)(so * ftanh(c));
      size_t oidx = (size_t)(m0 + rbase + v) * Hc + (j0 + col);
      hdst[oidx] = hh;
      yout[(size_t)t * Bc * Hc + oidx] = hh;
    }
    ++gen; gsync(cnt, gen * NWG);
  }

  // Persist final cell state for the decoder (h final already in hpp[0]).
#pragma unroll
  for (int v = 0; v < 8; ++v)
    cbuf[(size_t)(m0 + rbase + v) * Hc + (j0 + col)] = creg[v];
}

// ---------------------------------------------------------------------------
// In-place online softmax over V=16000: fused max+sum pass (running rescale),
// wave32 __shfl_xor reduction + cross-wave LDS combine, then normalize pass.
// ---------------------------------------------------------------------------
__global__ void softmax_rows_kernel(float* __restrict__ out, int cols) {
  float* p = out + (size_t)blockIdx.x * cols;
  float m = -3.402823466e38f, s = 0.0f;
  for (int i = threadIdx.x; i < cols; i += 256) {
    float x = p[i];
    if (x > m) { s = s * __expf(m - x) + 1.0f; m = x; }
    else        s += __expf(x - m);
  }
#pragma unroll
  for (int o = 16; o > 0; o >>= 1) {
    float mo = __shfl_xor(m, o, 32);
    float so = __shfl_xor(s, o, 32);
    float M = fmaxf(m, mo);
    s = s * __expf(m - M) + so * __expf(mo - M);
    m = M;
  }
  __shared__ float sm[8], ss[8];
  int wave = threadIdx.x >> 5, lane = threadIdx.x & 31;
  if (lane == 0) { sm[wave] = m; ss[wave] = s; }
  __syncthreads();
  if (threadIdx.x == 0) {
    float M = sm[0], S = ss[0];
#pragma unroll
    for (int i = 1; i < 8; ++i) {
      float M2 = fmaxf(M, sm[i]);
      S = S * __expf(M - M2) + ss[i] * __expf(sm[i] - M2);
      M = M2;
    }
    sm[0] = M; ss[0] = 1.0f / S;
  }
  __syncthreads();
  float M = sm[0], inv = ss[0];
  for (int i = threadIdx.x; i < cols; i += 256)
    p[i] = __expf(p[i] - M) * inv;
}

// ---------------------------------------------------------------------------
// Host orchestration.  Workspace budget ~115 MB.
// ---------------------------------------------------------------------------
extern "C" void kernel_launch(void* const* d_in, const int* in_sizes, int n_in,
                              void* d_out, int out_size, void* d_ws,
                              size_t ws_size, hipStream_t stream) {
  (void)in_sizes; (void)n_in; (void)out_size; (void)ws_size;
  const int*   batch = (const int*)d_in[0];
  const float* feats = (const float*)d_in[1];
  const float* Wih[4] = {(const float*)d_in[2],  (const float*)d_in[6],
                         (const float*)d_in[10], (const float*)d_in[14]};
  const float* Whh[4] = {(const float*)d_in[3],  (const float*)d_in[7],
                         (const float*)d_in[11], (const float*)d_in[15]};
  const float* bih[4] = {(const float*)d_in[4],  (const float*)d_in[8],
                         (const float*)d_in[12], (const float*)d_in[16]};
  const float* bhh[4] = {(const float*)d_in[5],  (const float*)d_in[9],
                         (const float*)d_in[13], (const float*)d_in[17]};
  const float* linW = (const float*)d_in[18];
  const float* linb = (const float*)d_in[19];
  float* out = (float*)d_out;

  char* w = (char*)d_ws;
  size_t off = 0;
  auto alloc = [&](size_t bytes) -> char* {
    char* p = w + off;
    off = (off + bytes + 255) & ~(size_t)255;
    return p;
  };

  // Zeroed each call: barrier counters (state init handled in-kernel).
  unsigned* counters = (unsigned*)alloc(4 * sizeof(unsigned));
  size_t zeroBytes = off;

  float*    cstate[2] = {(float*)alloc(Bc * Hc * 4), (float*)alloc(Bc * Hc * 4)};
  _Float16* hpair[2]  = {(_Float16*)alloc(2 * Bc * Hc * 2),
                         (_Float16*)alloc(2 * Bc * Hc * 2)};

  _Float16* pWih[4]; _Float16* pWhh[4]; float* bsum[4];
  const int Kih[4] = {Dc, Hc, Dc, Hc};
  for (int l = 0; l < 4; ++l) {
    pWih[l] = (_Float16*)alloc((size_t)N4 * Kih[l] * 2);
    pWhh[l] = (_Float16*)alloc((size_t)N4 * Hc * 2);
    bsum[l] = (float*)alloc(N4 * 4);
  }
  _Float16* pLin = (_Float16*)alloc((size_t)Vc * Hc * 2);
  _Float16* x16  = (_Float16*)alloc((size_t)MBT * Dc * 2);
  _Float16* yA   = (_Float16*)alloc((size_t)MBT * Hc * 2);
  _Float16* yB   = (_Float16*)alloc((size_t)MBT * Hc * 2);
  float*    xg   = (float*)alloc((size_t)MBT * N4 * 4);

  hipMemsetAsync(d_ws, 0, zeroBytes, stream);

  // Pack weights to WMMA fragment layout (f16), combine biases.
  auto pack = [&](const float* src, _Float16* dst, int N, int K) {
    int total = N * K;
    pack_w_kernel<<<(total + 255) / 256, 256, 0, stream>>>(src, dst, N, K);
  };
  for (int l = 0; l < 4; ++l) {
    pack(Wih[l], pWih[l], N4, Kih[l]);
    pack(Whh[l], pWhh[l], N4, Hc);
    add_bias_kernel<<<(N4 + 255) / 256, 256, 0, stream>>>(bih[l], bhh[l],
                                                          bsum[l], N4);
  }
  pack(linW, pLin, Vc, Hc);

  embed_cast_kernel<<<(Tc * Bc * Dc + 255) / 256, 256, 0, stream>>>(batch,
                                                                    feats, x16);

  // One layer = input-projection GEMM + persistent fused recurrent scan.
  auto run_layer = [&](const _Float16* Ain, int lda, int K, int l, int state,
                       _Float16* yout, int initZero, int cidx) {
    wmma_gemm_bias_kernel<<<dim3(N4 / 64, MBT / 256), 256, 0, stream>>>(
        Ain, lda, pWih[l], bsum[l], xg, N4, K, 0);
    lstm_scan_kernel<<<32, 128, 0, stream>>>(xg, pWhh[l], cstate[state],
                                             hpair[state], yout, initZero,
                                             counters + cidx);
  };

  run_layer(x16, Dc, Dc, 0, 0, yA, /*initZero=*/1, 0);  // enc layer 0
  run_layer(yA, Hc, Hc, 1, 1, yB, /*initZero=*/1, 1);   // enc layer 1
  run_layer(x16, Dc, Dc, 2, 0, yA, /*initZero=*/0, 2);  // dec layer 0 (enc state)
  run_layer(yA, Hc, Hc, 3, 1, yB, /*initZero=*/0, 3);   // dec layer 1 (enc state)

  // Logits GEMM (time-major rows scattered to [B][T][V]) + in-place softmax.
  wmma_gemm_bias_kernel<<<dim3(Vc / 64, MBT / 256), 256, 0, stream>>>(
      yB, Hc, pLin, linb, out, Vc, Hc, /*remap=*/1);
  softmax_rows_kernel<<<MBT, 256, 0, stream>>>(out, Vc);
}
